// BiLSTM_CRF_PARALLEL_60258391163094
// MI455X (gfx1250) — matrix operand order repaired and checked by reading, not compiled, not run on hardware
//
#include <hip/hip_runtime.h>
#include <hip/hip_bf16.h>
#include <stdint.h>

// ---------------------------------------------------------------------------
// Problem constants (from reference)
// ---------------------------------------------------------------------------
#define Vv   50000
#define Ee   256
#define Hh   512
#define H2   256
#define Bb   128
#define Ll   512
#define Tt   9
#define TAG_START 7
#define TAG_STOP  8
#define NEGINF (-10000.0f)

typedef _Float16 f16;
typedef __attribute__((ext_vector_type(16))) _Float16 v16h;
typedef __attribute__((ext_vector_type(8)))  float    v8f;

union V16 { uint4 u[2]; v16h v; };

#if defined(__has_builtin)
#if __has_builtin(__builtin_amdgcn_sched_barrier)
#define SCHED_FENCE() __builtin_amdgcn_sched_barrier(0)
#endif
#endif
#ifndef SCHED_FENCE
#define SCHED_FENCE()
#endif

// LDS layout for the LSTM kernel (dynamic shared):
//   f16  hbuf0[128][HSTR]   66 KB   h double buffer A
//   f16  hbuf1[128][HSTR]   66 KB   h double buffer B
//   f32  cbuf [128][CSTR]  132 KB   persistent cell state (thread-private slots)
//   f32  biasb[1024]         4 KB   bih+bhh
// total 274,432 B  (WGP LDS = 320 KB)
// HSTR/CSTR = 264: 132-dword rows -> bank stride 4, conflict-free tile access.
#define HSTR 264
#define CSTR 264
#define LSTM_SMEM (2 * Bb * HSTR * 2 + Bb * CSTR * 4 + 1024 * 4)

__device__ __forceinline__ float sigmoidf_(float x) {
  return 1.0f / (1.0f + __expf(-x));
}
__device__ __forceinline__ float tanhf_(float x) {
  float e = __expf(2.0f * x);
  return 1.0f - 2.0f / (e + 1.0f);
}

// ---------------------------------------------------------------------------
// Kernel 1: embedding gather + cast to f16  (x_h[b][l][e])
// ---------------------------------------------------------------------------
__global__ void k_pack_x(const int* __restrict__ tokens,
                         const float* __restrict__ emb,
                         f16* __restrict__ xh) {
  int idx = blockIdx.x * blockDim.x + threadIdx.x;
  const int N = Bb * Ll * Ee;
  if (idx >= N) return;
  int e   = idx & (Ee - 1);
  int bl  = idx >> 8;            // Ee = 256
  int tok = tokens[bl];
  xh[idx] = (f16)emb[(size_t)tok * Ee + e];
}

// ---------------------------------------------------------------------------
// Kernel 2: pack combined weights Wc[dir][n (0..1023)][k (0..511)] f16,
// k<256 -> Wih[n][k], k>=256 -> Whh[n][k-256]. Row-major [N][K] so a WMMA
// B-tile load mirrors an A-tile load with "row = output column".
// ---------------------------------------------------------------------------
__global__ void k_pack_w(const float* __restrict__ Wih_f,
                         const float* __restrict__ Whh_f,
                         const float* __restrict__ Wih_b,
                         const float* __restrict__ Whh_b,
                         f16* __restrict__ Wc) {
  int idx = blockIdx.x * blockDim.x + threadIdx.x;
  const int N = 2 * 1024 * 512;
  if (idx >= N) return;
  int dir = idx >> 19;
  int rem = idx & ((1 << 19) - 1);
  int n = rem >> 9;
  int k = rem & 511;
  const float* Wih = dir ? Wih_b : Wih_f;
  const float* Whh = dir ? Whh_b : Whh_f;
  float v = (k < H2) ? Wih[n * H2 + k] : Whh[n * H2 + (k - H2)];
  Wc[idx] = (f16)v;
}

// ---------------------------------------------------------------------------
// Kernel 3: persistent bidirectional LSTM. blockIdx.x = direction (0 fwd, 1 bwd).
// 1024 threads = 32 waves. Wave w: mTile = w&7, nGroup = w>>3 covers nTiles
// nGroup*4..+3 of H2/16=16. Tile-sets processed SEQUENTIALLY in a forced-
// rolled loop (only 32 accumulator VGPRs live; ~100 total, under the tight
// per-wave cap at 32 waves/WGP). All per-thread state that needs dynamic
// indexing (cell state, bias) lives in LDS, not private arrays -> zero
// scratch. h is double-buffered in LDS -> one barrier per step. The x A-tile
// address gets an opaque zero offset per tile-set so LICM cannot hoist (and
// then spill) the loop-invariant global loads, while keeping pointer
// provenance (global_load, not flat_load).
// Per step: z = [x_t | h] (M=128,K=512) x Wc^T (K=512,N=1024) via
// v_wmma_f32_16x16x32_f16, then elementwise gates.
// ---------------------------------------------------------------------------
__global__ void __launch_bounds__(1024)
k_lstm(const f16* __restrict__ xh,       // [B][L][E] f16
       const f16* __restrict__ Wc,       // [2][1024][512] f16
       const float* __restrict__ bih_f, const float* __restrict__ bhh_f,
       const float* __restrict__ bih_b, const float* __restrict__ bhh_b,
       const float* __restrict__ h0,    // [2][B][H2]
       const float* __restrict__ c0,    // [2][B][H2]
       float* __restrict__ hout)        // [B][L][H]
{
  extern __shared__ char smem[];
  f16*   hbuf0 = (f16*)smem;                          // [Bb][HSTR]
  f16*   hbuf1 = hbuf0 + Bb * HSTR;                   // [Bb][HSTR]
  float* cbuf  = (float*)(smem + 2 * Bb * HSTR * 2);  // [Bb][CSTR]
  float* biasb = cbuf + Bb * CSTR;                    // [1024]

  const int dir  = blockIdx.x;
  const int tid  = threadIdx.x;
  const int lane = tid & 31;
  const int w    = tid >> 5;
  const int half = lane >> 4;            // 0: lanes 0-15, 1: lanes 16-31
  const int l15  = lane & 15;
  const int mTile  = w & 7;
  const int nGroup = w >> 3;

  const float* bih = dir ? bih_b : bih_f;
  const float* bhh = dir ? bhh_b : bhh_f;
  const f16*   Wcd = Wc + (size_t)dir * 1024 * 512;
  const int dirOff = dir * H2;

  // --- init h (LDS buffer 0) and c (LDS) from h0/c0; bias table ---
  for (int i = tid; i < Bb * H2; i += 1024) {
    int b = i >> 8, n = i & 255;
    hbuf0[b * HSTR + n] = (f16)h0[(size_t)(dir * Bb + b) * H2 + n];
    cbuf [b * CSTR + n] =       c0[(size_t)(dir * Bb + b) * H2 + n];
  }
  if (tid < 1024) biasb[tid] = bih[tid] + bhh[tid];
  __syncthreads();

  const int browA      = mTile * 16 + l15;        // A-matrix row for this lane
  const int browC_base = mTile * 16 + half * 8;   // C/D row base for VGPR r

  for (int step = 0; step < Ll; ++step) {
    const int t = dir ? (Ll - 1 - step) : step;
    const f16* hOld = (step & 1) ? hbuf1 : hbuf0;
    f16*       hNew = (step & 1) ? hbuf0 : hbuf1;
    const f16* xrow = xh + ((size_t)browA * Ll + t) * Ee;

    // prefetch next step's x row into cache while this step computes
    if (step + 1 < Ll) {
      const int tn = dir ? (Ll - 2 - step) : (step + 1);
      const char* nrow = (const char*)(xh + ((size_t)browA * Ll + tn) * Ee);
      __builtin_prefetch(nrow, 0, 0);
      __builtin_prefetch(nrow + 128, 0, 0);
      __builtin_prefetch(nrow + 256, 0, 0);
      __builtin_prefetch(nrow + 384, 0, 0);
    }

    #pragma unroll 1   // keep rolled: bounds register pressure
    for (int ts = 0; ts < 4; ++ts) {
      const int nTile = nGroup * 4 + ts;
      const int ncolH = nTile * 16 + l15;

      // Opaque zero offset: re-executed every iteration so the (ts-invariant)
      // x A-tile addresses are not loop-invariant -> LICM cannot hoist and
      // spill them. Pointer provenance is preserved -> global_load_b128.
      int zoff = 0;
      asm volatile("" : "+v"(zoff));
      const f16* xr = xrow + zoff;

      // accumulators start at bias (bias depends only on column)
      v8f acc[4];
      #pragma unroll
      for (int g = 0; g < 4; ++g) {
        float bv = biasb[g * H2 + ncolH];
        #pragma unroll
        for (int r = 0; r < 8; ++r) acc[g][r] = bv;
      }

      // x contribution: A from global (L0-resident row), B streamed from L2.
      // A tile 16x32 f16: lane holds row l15,
      // K = kb*32 + half*16 + {0..15}
      #pragma unroll
      for (int kb = 0; kb < 8; ++kb) {
        V16 a;
        const uint4* p = (const uint4*)(xr + kb * 32 + half * 16);
        a.u[0] = p[0];
        a.u[1] = p[1];
        #pragma unroll
        for (int g = 0; g < 4; ++g) {
          // B tile 32x16: lane holds column ncol, K = kb*32 + half*16 + {0..15}
          int ncol = g * H2 + ncolH;
          const uint4* q = (const uint4*)(Wcd +
              ((size_t)ncol * 512 + kb * 32 + half * 16));
          V16 bmat;
          bmat.u[0] = q[0];
          bmat.u[1] = q[1];
          acc[g] = __builtin_amdgcn_wmma_f32_16x16x32_f16(
              false, a.v, false, bmat.v, (short)0, acc[g], false, false);
        }
      }

      // h contribution: A from LDS (double-buffered h), B streamed from L2
      #pragma unroll
      for (int kb = 0; kb < 8; ++kb) {
        V16 a;
        const uint4* p = (const uint4*)(&hOld[(size_t)browA * HSTR +
                                             kb * 32 + half * 16]);
        a.u[0] = p[0];
        a.u[1] = p[1];
        #pragma unroll
        for (int g = 0; g < 4; ++g) {
          int ncol = g * H2 + ncolH;
          const uint4* q = (const uint4*)(Wcd +
              ((size_t)ncol * 512 + (kb + 8) * 32 + half * 16));
          V16 bmat;
          bmat.u[0] = q[0];
          bmat.u[1] = q[1];
          acc[g] = __builtin_amdgcn_wmma_f32_16x16x32_f16(
              false, a.v, false, bmat.v, (short)0, acc[g], false, false);
        }
      }

      // elementwise gates: c = sig(f)*c + sig(i)*tanh(g); h = sig(o)*tanh(c)
      // c lives in LDS; each element is owned by exactly this thread.
      #pragma unroll
      for (int r = 0; r < 8; ++r) {
        int brow = browC_base + r;
        float ig = sigmoidf_(acc[0][r]);
        float fg = sigmoidf_(acc[1][r]);
        float gg = tanhf_(acc[2][r]);
        float og = sigmoidf_(acc[3][r]);
        float c  = fg * cbuf[brow * CSTR + ncolH] + ig * gg;
        cbuf[brow * CSTR + ncolH] = c;
        float hv = og * tanhf_(c);
        hNew[brow * HSTR + ncolH] = (f16)hv;
        hout[((size_t)brow * Ll + t) * Hh + dirOff + ncolH] = hv;
      }
      SCHED_FENCE();   // keep tile-set live ranges from merging
    }
    __syncthreads();   // h_new complete and visible before next step reads it
  }
}

// ---------------------------------------------------------------------------
// Kernel 4: feats = lstm_out @ W_out^T + b_out. One wave per (b,l) row.
// ---------------------------------------------------------------------------
__global__ void k_feats(const float* __restrict__ hout,   // [B][L][H]
                        const float* __restrict__ Wout,   // [T][H]
                        const float* __restrict__ bout,   // [T]
                        float* __restrict__ feats)        // [B][L][T]
{
  __shared__ float wsh[Tt * Hh];
  __shared__ float bsh[Tt];
  for (int i = threadIdx.x; i < Tt * Hh; i += blockDim.x) wsh[i] = Wout[i];
  if (threadIdx.x < Tt) bsh[threadIdx.x] = bout[threadIdx.x];
  __syncthreads();

  int lane = threadIdx.x & 31;
  int gw   = blockIdx.x * (blockDim.x >> 5) + (threadIdx.x >> 5);  // (b,l) id
  if (gw >= Bb * Ll) return;

  const float* row = hout + (size_t)gw * Hh;
  float xv[16];
  #pragma unroll
  for (int k = 0; k < 16; ++k) xv[k] = row[lane + k * 32];

  for (int tg = 0; tg < Tt; ++tg) {
    float p = 0.0f;
    #pragma unroll
    for (int k = 0; k < 16; ++k) p += xv[k] * wsh[tg * Hh + lane + k * 32];
    #pragma unroll
    for (int off = 16; off > 0; off >>= 1) p += __shfl_xor(p, off, 32);
    if (lane == 0) feats[(size_t)gw * Tt + tg] = p + bsh[tg];
  }
}

// ---------------------------------------------------------------------------
// Kernel 5: Viterbi. One thread per batch row (128 threads, 1 block).
// trans[i][j] = score j -> i. First-index argmax (strict >) matches jnp.
// ---------------------------------------------------------------------------
__global__ void k_viterbi(const float* __restrict__ feats,  // [B][L][T]
                          const float* __restrict__ trans,  // [T][T]
                          unsigned char* __restrict__ bptr, // [L][B][T]
                          float* __restrict__ out)          // [B] ++ [B][L]
{
  __shared__ float tr[Tt * Tt];
  for (int i = threadIdx.x; i < Tt * Tt; i += blockDim.x) tr[i] = trans[i];
  __syncthreads();

  int b = threadIdx.x;
  if (b >= Bb) return;

  float alpha[Tt];
  #pragma unroll
  for (int j = 0; j < Tt; ++j) alpha[j] = NEGINF;
  alpha[TAG_START] = 0.0f;

  for (int t = 0; t < Ll; ++t) {
    const float* ft = feats + ((size_t)b * Ll + t) * Tt;
    float na[Tt];
    #pragma unroll
    for (int nxt = 0; nxt < Tt; ++nxt) {
      float best = alpha[0] + tr[nxt * Tt + 0];
      int   barg = 0;
      #pragma unroll
      for (int prev = 1; prev < Tt; ++prev) {
        float s = alpha[prev] + tr[nxt * Tt + prev];
        if (s > best) { best = s; barg = prev; }
      }
      na[nxt] = best + ft[nxt];
      bptr[((size_t)t * Bb + b) * Tt + nxt] = (unsigned char)barg;
    }
    #pragma unroll
    for (int j = 0; j < Tt; ++j) alpha[j] = na[j];
  }

  float best = alpha[0] + tr[TAG_STOP * Tt + 0];
  int   tag  = 0;
  #pragma unroll
  for (int j = 1; j < Tt; ++j) {
    float s = alpha[j] + tr[TAG_STOP * Tt + j];
    if (s > best) { best = s; tag = j; }
  }
  out[b] = best;

  float* path = out + Bb + (size_t)b * Ll;
  path[Ll - 1] = (float)tag;
  for (int t = Ll - 1; t >= 1; --t) {
    tag = (int)bptr[((size_t)t * Bb + b) * Tt + tag];
    path[t - 1] = (float)tag;
  }
}

// ---------------------------------------------------------------------------
// Host-side launcher
// ---------------------------------------------------------------------------
extern "C" void kernel_launch(void* const* d_in, const int* in_sizes, int n_in,
                              void* d_out, int out_size, void* d_ws, size_t ws_size,
                              hipStream_t stream) {
  (void)in_sizes; (void)n_in; (void)out_size; (void)ws_size;

  const int*   tokens = (const int*)  d_in[0];
  const float* emb    = (const float*)d_in[1];
  const float* Wih_f  = (const float*)d_in[2];
  const float* Whh_f  = (const float*)d_in[3];
  const float* bih_f  = (const float*)d_in[4];
  const float* bhh_f  = (const float*)d_in[5];
  const float* Wih_b  = (const float*)d_in[6];
  const float* Whh_b  = (const float*)d_in[7];
  const float* bih_b  = (const float*)d_in[8];
  const float* bhh_b  = (const float*)d_in[9];
  const float* Wout   = (const float*)d_in[10];
  const float* bout   = (const float*)d_in[11];
  const float* trans  = (const float*)d_in[12];
  const float* h0     = (const float*)d_in[13];
  const float* c0     = (const float*)d_in[14];
  float* out = (float*)d_out;

  // workspace carve-up (regions naturally 256-byte aligned)
  char* ws = (char*)d_ws;
  size_t off = 0;
  f16* xh = (f16*)(ws + off);                 off += (size_t)Bb * Ll * Ee * 2;   // 32 MiB
  f16* Wc = (f16*)(ws + off);                 off += (size_t)2 * 1024 * 512 * 2; // 2 MiB
  float* hout  = (float*)(ws + off);          off += (size_t)Bb * Ll * Hh * 4;   // 128 MiB
  float* feats = (float*)(ws + off);          off += (size_t)Bb * Ll * Tt * 4;
  unsigned char* bptr = (unsigned char*)(ws + off);

  // 1) embedding gather -> f16
  {
    int N = Bb * Ll * Ee;
    k_pack_x<<<(N + 255) / 256, 256, 0, stream>>>(tokens, emb, xh);
  }
  // 2) weight pack -> f16 [dir][1024][512]
  {
    int N = 2 * 1024 * 512;
    k_pack_w<<<(N + 255) / 256, 256, 0, stream>>>(Wih_f, Whh_f, Wih_b, Whh_b, Wc);
  }
  // 3) persistent LSTM, one WG per direction; 268 KiB dynamic LDS
  hipFuncSetAttribute((const void*)k_lstm,
                      hipFuncAttributeMaxDynamicSharedMemorySize, LSTM_SMEM);
  k_lstm<<<2, 1024, LSTM_SMEM, stream>>>(xh, Wc, bih_f, bhh_f, bih_b, bhh_b,
                                         h0, c0, hout);
  // 4) output projection
  {
    int waves_per_block = 256 / 32;
    int nblocks = (Bb * Ll) / waves_per_block;
    k_feats<<<nblocks, 256, 0, stream>>>(hout, Wout, bout, feats);
  }
  // 5) Viterbi
  k_viterbi<<<1, 128, 0, stream>>>(feats, trans, bptr, out);
}
